// DVMNet_59107339927989
// MI455X (gfx1250) — matrix operand order repaired and checked by reading, not compiled, not run on hardware
//
#include <hip/hip_runtime.h>
#include <hip/hip_bf16.h>

typedef __attribute__((ext_vector_type(16))) _Float16 v16h;
typedef __attribute__((ext_vector_type(8)))  float    v8f;

#define N_TOT   804
#define N_LNC   284
#define N_MI    520
#define E_GCN   20000
#define E_LBL   600000
#define ALPHA_L 0.2f
#define BN_EPS  1e-5f

// CDNA5 async global->LDS path (ASYNCcnt), guarded so both toolchains compile.
#if defined(__has_builtin)
#if __has_builtin(__builtin_amdgcn_global_load_async_to_lds_b128) && \
    __has_builtin(__builtin_amdgcn_s_wait_asynccnt)
#define USE_ASYNC_LDS 1
#endif
#endif

#ifdef USE_ASYNC_LDS
#define AS_LOCAL  __attribute__((address_space(3)))
typedef int v4i __attribute__((vector_size(4 * sizeof(int))));
#endif

// ---------------------------------------------------------------------------
// Fused per-sample GAT (lnc): 64-node graph, 4 heads(12->30) + out(120->30),
// log_softmax, BN+ReLU over nodes, lnc1(30->1), lnc2(64->100), residual.
// One workgroup (64 threads, 2 waves) per sample. Writes x_new rows 0..283.
// ---------------------------------------------------------------------------
__global__ void gat_lnc_kernel(const float* __restrict__ x,        // [804,100]
                               const float* __restrict__ x_lnc,    // [284,64,64]
                               const float* __restrict__ node,     // [64,12]
                               const float* __restrict__ WA,       // [4,12,30]
                               const float* __restrict__ aA,       // [4,60]
                               const float* __restrict__ Wo,       // [120,30]
                               const float* __restrict__ ao,       // [60]
                               const float* __restrict__ bng,      // [30]
                               const float* __restrict__ bnb,      // [30]
                               const float* __restrict__ l1w,      // [30]
                               const float* __restrict__ l1b,      // [1]
                               const float* __restrict__ l2w,      // [64,100]
                               const float* __restrict__ l2b,      // [100]
                               float* __restrict__ x_new)          // [804,100]
{
    const int n = threadIdx.x;   // node 0..63
    const int s = blockIdx.x;    // sample

    __shared__ float s_adj[64 * 64];
    __shared__ float s_wh[64 * 30];
    __shared__ float s_f2[64];
    __shared__ float s_heads[64 * 120];
    __shared__ float s_bnm[30], s_bns[30];
    __shared__ float s_s1[64];

    for (int t = n; t < 64 * 64; t += 64) {
        float a = x_lnc[(size_t)s * 4096 + t];
        s_adj[t] = (a > 0.03f) ? a : 0.0f;
    }
    __syncthreads();

    // ------------------ 4 attention heads ------------------
    for (int h = 0; h < 4; ++h) {
        __syncthreads();  // previous head's Wh fully consumed
        float acc[30];
        #pragma unroll
        for (int f = 0; f < 30; ++f) acc[f] = 0.0f;
        #pragma unroll
        for (int k = 0; k < 12; ++k) {
            float xv = node[n * 12 + k];
            const float* wr = WA + (h * 12 + k) * 30;
            #pragma unroll
            for (int f = 0; f < 30; ++f) acc[f] += xv * wr[f];
        }
        float f1 = 0.0f, f2 = 0.0f;
        const float* a1 = aA + h * 60;
        #pragma unroll
        for (int f = 0; f < 30; ++f) {
            s_wh[n * 30 + f] = acc[f];
            f1 += acc[f] * a1[f];
            f2 += acc[f] * a1[30 + f];
        }
        s_f2[n] = f2;
        __syncthreads();

        float mx = -3.0e38f;
        for (int j = 0; j < 64; ++j) {
            float e = f1 + s_f2[j];
            e = (e > 0.0f) ? e : ALPHA_L * e;
            e = (s_adj[n * 64 + j] > 0.0f) ? e : -9.0e15f;
            mx = fmaxf(mx, e);
        }
        float sum = 0.0f;
        float oacc[30];
        #pragma unroll
        for (int f = 0; f < 30; ++f) oacc[f] = 0.0f;
        for (int j = 0; j < 64; ++j) {
            float e = f1 + s_f2[j];
            e = (e > 0.0f) ? e : ALPHA_L * e;
            e = (s_adj[n * 64 + j] > 0.0f) ? e : -9.0e15f;
            float w = __expf(e - mx);
            sum += w;
            const float* whj = &s_wh[j * 30];
            #pragma unroll
            for (int f = 0; f < 30; ++f) oacc[f] += w * whj[f];
        }
        float inv = 1.0f / sum;
        #pragma unroll
        for (int f = 0; f < 30; ++f) {
            float v = oacc[f] * inv;
            s_heads[n * 120 + h * 30 + f] = (v > 0.0f) ? v : (__expf(v) - 1.0f); // ELU
        }
    }
    __syncthreads();

    // ------------------ output attention layer (120 -> 30) ------------------
    {
        float acc[30];
        #pragma unroll
        for (int f = 0; f < 30; ++f) acc[f] = 0.0f;
        for (int c = 0; c < 120; ++c) {
            float hv = s_heads[n * 120 + c];
            const float* wr = Wo + c * 30;
            #pragma unroll
            for (int f = 0; f < 30; ++f) acc[f] += hv * wr[f];
        }
        float f1 = 0.0f, f2 = 0.0f;
        #pragma unroll
        for (int f = 0; f < 30; ++f) {
            s_wh[n * 30 + f] = acc[f];
            f1 += acc[f] * ao[f];
            f2 += acc[f] * ao[30 + f];
        }
        s_f2[n] = f2;
        __syncthreads();

        float mx = -3.0e38f;
        for (int j = 0; j < 64; ++j) {
            float e = f1 + s_f2[j];
            e = (e > 0.0f) ? e : ALPHA_L * e;
            e = (s_adj[n * 64 + j] > 0.0f) ? e : -9.0e15f;
            mx = fmaxf(mx, e);
        }
        float sum = 0.0f;
        float oacc[30];
        #pragma unroll
        for (int f = 0; f < 30; ++f) oacc[f] = 0.0f;
        for (int j = 0; j < 64; ++j) {
            float e = f1 + s_f2[j];
            e = (e > 0.0f) ? e : ALPHA_L * e;
            e = (s_adj[n * 64 + j] > 0.0f) ? e : -9.0e15f;
            float w = __expf(e - mx);
            sum += w;
            const float* whj = &s_wh[j * 30];
            #pragma unroll
            for (int f = 0; f < 30; ++f) oacc[f] += w * whj[f];
        }
        float inv = 1.0f / sum;
        float v[30], vmax = -3.0e38f;
        #pragma unroll
        for (int f = 0; f < 30; ++f) {
            float t = oacc[f] * inv;
            v[f] = (t > 0.0f) ? t : (__expf(t) - 1.0f);
            vmax = fmaxf(vmax, v[f]);
        }
        float lse = 0.0f;
        #pragma unroll
        for (int f = 0; f < 30; ++f) lse += __expf(v[f] - vmax);
        lse = vmax + __logf(lse);
        __syncthreads();          // all reads of s_wh done; reuse for g
        #pragma unroll
        for (int f = 0; f < 30; ++f) s_wh[n * 30 + f] = v[f] - lse;
    }
    __syncthreads();

    // BN over node axis (per-feature mean/var over 64 nodes)
    if (n < 30) {
        float m = 0.0f, q = 0.0f;
        for (int j = 0; j < 64; ++j) {
            float g = s_wh[j * 30 + n];
            m += g; q += g * g;
        }
        m *= (1.0f / 64.0f); q *= (1.0f / 64.0f);
        float var = fmaxf(q - m * m, 0.0f);
        s_bnm[n] = m;
        s_bns[n] = rsqrtf(var + BN_EPS) * bng[n];
    }
    __syncthreads();

    // lnc1: [30] -> scalar per node (BN+ReLU inline)
    {
        float acc = l1b[0];
        #pragma unroll
        for (int f = 0; f < 30; ++f) {
            float y = (s_wh[n * 30 + f] - s_bnm[f]) * s_bns[f] + bnb[f];
            acc += fmaxf(y, 0.0f) * l1w[f];
        }
        s_s1[n] = acc;
    }
    __syncthreads();

    // lnc2: [64] -> [100], residual into x_new row s
    for (int j = n; j < 100; j += 64) {
        float acc = l2b[j];
        for (int m = 0; m < 64; ++m) acc += s_s1[m] * l2w[m * 100 + j];
        x_new[(size_t)s * 100 + j] = x[(size_t)s * 100 + j] + 0.2f * fmaxf(acc, 0.0f);
    }
}

// ---------------------------------------------------------------------------
// Fused per-sample GAT (mi): 16-node graphs. One wave (32 threads) / sample.
// ---------------------------------------------------------------------------
__global__ void gat_mi_kernel(const float* __restrict__ x,
                              const float* __restrict__ x_mi,     // [520,16,16]
                              const float* __restrict__ node,     // [16,8]
                              const float* __restrict__ WB,       // [4,8,30]
                              const float* __restrict__ aB,       // [4,60]
                              const float* __restrict__ Wo,       // [120,30]
                              const float* __restrict__ ao,       // [60]
                              const float* __restrict__ bng,
                              const float* __restrict__ bnb,
                              const float* __restrict__ m1w,      // [30]
                              const float* __restrict__ m1b,      // [1]
                              const float* __restrict__ m2w,      // [16,100]
                              const float* __restrict__ m2b,      // [100]
                              float* __restrict__ x_new)
{
    const int n = threadIdx.x;   // 0..31, nodes are n<16
    const int s = blockIdx.x;

    __shared__ float s_adj[16 * 16];
    __shared__ float s_wh[16 * 30];
    __shared__ float s_f2[16];
    __shared__ float s_heads[16 * 120];
    __shared__ float s_bnm[30], s_bns[30];
    __shared__ float s_s1[16];

    for (int t = n; t < 256; t += 32) {
        float a = x_mi[(size_t)s * 256 + t];
        s_adj[t] = (a > 0.03f) ? a : 0.0f;
    }
    __syncthreads();

    for (int h = 0; h < 4; ++h) {
        __syncthreads();
        if (n < 16) {
            float acc[30];
            #pragma unroll
            for (int f = 0; f < 30; ++f) acc[f] = 0.0f;
            #pragma unroll
            for (int k = 0; k < 8; ++k) {
                float xv = node[n * 8 + k];
                const float* wr = WB + (h * 8 + k) * 30;
                #pragma unroll
                for (int f = 0; f < 30; ++f) acc[f] += xv * wr[f];
            }
            float f2 = 0.0f;
            const float* a1 = aB + h * 60;
            #pragma unroll
            for (int f = 0; f < 30; ++f) {
                s_wh[n * 30 + f] = acc[f];
                f2 += acc[f] * a1[30 + f];
            }
            s_f2[n] = f2;
        }
        __syncthreads();
        if (n < 16) {
            const float* a1 = aB + h * 60;
            float f1 = 0.0f;
            #pragma unroll
            for (int f = 0; f < 30; ++f) f1 += s_wh[n * 30 + f] * a1[f];
            float mx = -3.0e38f;
            for (int j = 0; j < 16; ++j) {
                float e = f1 + s_f2[j];
                e = (e > 0.0f) ? e : ALPHA_L * e;
                e = (s_adj[n * 16 + j] > 0.0f) ? e : -9.0e15f;
                mx = fmaxf(mx, e);
            }
            float sum = 0.0f, oacc[30];
            #pragma unroll
            for (int f = 0; f < 30; ++f) oacc[f] = 0.0f;
            for (int j = 0; j < 16; ++j) {
                float e = f1 + s_f2[j];
                e = (e > 0.0f) ? e : ALPHA_L * e;
                e = (s_adj[n * 16 + j] > 0.0f) ? e : -9.0e15f;
                float w = __expf(e - mx);
                sum += w;
                const float* whj = &s_wh[j * 30];
                #pragma unroll
                for (int f = 0; f < 30; ++f) oacc[f] += w * whj[f];
            }
            float inv = 1.0f / sum;
            #pragma unroll
            for (int f = 0; f < 30; ++f) {
                float v = oacc[f] * inv;
                s_heads[n * 120 + h * 30 + f] = (v > 0.0f) ? v : (__expf(v) - 1.0f);
            }
        }
    }
    __syncthreads();

    if (n < 16) {
        float acc[30];
        #pragma unroll
        for (int f = 0; f < 30; ++f) acc[f] = 0.0f;
        for (int c = 0; c < 120; ++c) {
            float hv = s_heads[n * 120 + c];
            const float* wr = Wo + c * 30;
            #pragma unroll
            for (int f = 0; f < 30; ++f) acc[f] += hv * wr[f];
        }
        float f2 = 0.0f;
        #pragma unroll
        for (int f = 0; f < 30; ++f) {
            s_wh[n * 30 + f] = acc[f];
            f2 += acc[f] * ao[30 + f];
        }
        s_f2[n] = f2;
    }
    __syncthreads();
    if (n < 16) {
        float f1 = 0.0f;
        #pragma unroll
        for (int f = 0; f < 30; ++f) f1 += s_wh[n * 30 + f] * ao[f];
        float mx = -3.0e38f;
        for (int j = 0; j < 16; ++j) {
            float e = f1 + s_f2[j];
            e = (e > 0.0f) ? e : ALPHA_L * e;
            e = (s_adj[n * 16 + j] > 0.0f) ? e : -9.0e15f;
            mx = fmaxf(mx, e);
        }
        float sum = 0.0f, oacc[30];
        #pragma unroll
        for (int f = 0; f < 30; ++f) oacc[f] = 0.0f;
        for (int j = 0; j < 16; ++j) {
            float e = f1 + s_f2[j];
            e = (e > 0.0f) ? e : ALPHA_L * e;
            e = (s_adj[n * 16 + j] > 0.0f) ? e : -9.0e15f;
            float w = __expf(e - mx);
            sum += w;
            const float* whj = &s_wh[j * 30];
            #pragma unroll
            for (int f = 0; f < 30; ++f) oacc[f] += w * whj[f];
        }
        float inv = 1.0f / sum;
        float v[30], vmax = -3.0e38f;
        #pragma unroll
        for (int f = 0; f < 30; ++f) {
            float t = oacc[f] * inv;
            v[f] = (t > 0.0f) ? t : (__expf(t) - 1.0f);
            vmax = fmaxf(vmax, v[f]);
        }
        float lse = 0.0f;
        #pragma unroll
        for (int f = 0; f < 30; ++f) lse += __expf(v[f] - vmax);
        lse = vmax + __logf(lse);
        #pragma unroll
        for (int f = 0; f < 30; ++f) s_wh[n * 30 + f] = v[f] - lse;
    }
    __syncthreads();

    if (n < 30) {
        float m = 0.0f, q = 0.0f;
        for (int j = 0; j < 16; ++j) {
            float g = s_wh[j * 30 + n];
            m += g; q += g * g;
        }
        m *= (1.0f / 16.0f); q *= (1.0f / 16.0f);
        float var = fmaxf(q - m * m, 0.0f);
        s_bnm[n] = m;
        s_bns[n] = rsqrtf(var + BN_EPS) * bng[n];
    }
    __syncthreads();

    if (n < 16) {
        float acc = m1b[0];
        #pragma unroll
        for (int f = 0; f < 30; ++f) {
            float y = (s_wh[n * 30 + f] - s_bnm[f]) * s_bns[f] + bnb[f];
            acc += fmaxf(y, 0.0f) * m1w[f];
        }
        s_s1[n] = acc;
    }
    __syncthreads();

    const int row = N_LNC + s;
    for (int j = n; j < 100; j += 32) {
        float acc = m2b[j];
        #pragma unroll
        for (int m = 0; m < 16; ++m) acc += s_s1[m] * m2w[m * 100 + j];
        x_new[(size_t)row * 100 + j] = x[(size_t)row * 100 + j] + 0.2f * fmaxf(acc, 0.0f);
    }
}

// ---------------------------------------------------------------------------
// f16 staging:  A16[Mpad,Kpad] = pad(A[M,K]);  B16T[N,Kpad] = pad(Bᵀ)
// ---------------------------------------------------------------------------
__global__ void stage_a_f16(const float* __restrict__ A, _Float16* __restrict__ A16,
                            int M, int K, int Kpad, int Mpad)
{
    int t = blockIdx.x * blockDim.x + threadIdx.x;
    int tot = Mpad * Kpad;
    if (t >= tot) return;
    int m = t / Kpad, k = t % Kpad;
    A16[t] = (m < M && k < K) ? (_Float16)A[(size_t)m * K + k] : (_Float16)0.0f;
}

__global__ void stage_bT_f16(const float* __restrict__ B, _Float16* __restrict__ BT,
                             int K, int N, int Kpad)
{
    int t = blockIdx.x * blockDim.x + threadIdx.x;
    int tot = N * Kpad;
    if (t >= tot) return;
    int n = t / Kpad, k = t % Kpad;
    BT[t] = (k < K) ? (_Float16)B[(size_t)k * N + n] : (_Float16)0.0f;
}

// ---------------------------------------------------------------------------
// WMMA GEMM: C[M,N] = A x B.  B (<=16KB f16) is staged once per workgroup
// into LDS via CDNA5 GLOBAL_LOAD_ASYNC_TO_LDS_B128 (ASYNCcnt) since every
// wave/tile reuses it; A fragments stream from global (b128 pairs).
// One wave per 16x16 C tile; K swept with v_wmma_f32_16x16x32_f16.
// VGPR layouts per CDNA5 ISA §7.12.2 (wave32).
// ---------------------------------------------------------------------------
__global__ void wmma_gemm_f16(const _Float16* __restrict__ A16,
                              const _Float16* __restrict__ B16T,
                              float* __restrict__ C,
                              int M, int N, int Kpad, int Mtiles, int Ntiles)
{
    __shared__ __align__(16) _Float16 s_b[64 * 128];   // 16 KB max (N=64, Kpad<=128)

    const int nhalves = N * Kpad;
    for (int c = threadIdx.x * 8; c < nhalves; c += blockDim.x * 8) {
#ifdef USE_ASYNC_LDS
        __builtin_amdgcn_global_load_async_to_lds_b128(
            (v4i*)(B16T + c), (AS_LOCAL v4i*)(&s_b[c]), 0, 0);
#else
        *reinterpret_cast<float4*>(&s_b[c]) =
            *reinterpret_cast<const float4*>(B16T + c);
#endif
    }
#ifdef USE_ASYNC_LDS
    __builtin_amdgcn_s_wait_asynccnt(0);
#endif
    __syncthreads();

    int wave = threadIdx.x >> 5;
    int lane = threadIdx.x & 31;
    int tile = blockIdx.x * 8 + wave;            // 8 waves per block
    if (tile >= Mtiles * Ntiles) return;
    int mt = tile / Ntiles;
    int nt = tile % Ntiles;
    int lm = lane & 15;
    bool hi = lane >= 16;

    v8f c = {};
    for (int k0 = 0; k0 < Kpad; k0 += 32) {
        // A fragment: lane = row m=lm; halves [0..7]=K(+0..7), [8..15]=K(+16..23)
        const _Float16* ap = A16 + (size_t)(mt * 16 + lm) * Kpad + k0 + (hi ? 8 : 0);
        v16h a;
        #pragma unroll
        for (int i = 0; i < 8; ++i) { a[i] = ap[i]; a[i + 8] = ap[i + 16]; }
        // B fragment from LDS: lane = col n=lm; halves [0..7]=K(+0..7), [8..15]=K(+16..23)
        const _Float16* bp = &s_b[(nt * 16 + lm) * Kpad + k0 + (hi ? 8 : 0)];
        v16h b;
        #pragma unroll
        for (int i = 0; i < 8; ++i) { b[i] = bp[i]; b[i + 8] = bp[i + 16]; }
        c = __builtin_amdgcn_wmma_f32_16x16x32_f16(false, a, false, b,
                                                   (short)0, c, false, false);
    }
    #pragma unroll
    for (int i = 0; i < 8; ++i) {
        int row = mt * 16 + (hi ? 8 + i : i);
        if (row < M) C[(size_t)row * N + nt * 16 + lm] = c[i];
    }
}

// ---------------------------------------------------------------------------
// GCN helpers
// ---------------------------------------------------------------------------
__global__ void deg_init_kernel(float* __restrict__ deg)
{
    int i = blockIdx.x * blockDim.x + threadIdx.x;
    if (i < N_TOT) deg[i] = 1.0f;   // self-loop
}

__global__ void deg_acc_kernel(const int* __restrict__ dst, float* __restrict__ deg)
{
    int e = blockIdx.x * blockDim.x + threadIdx.x;
    if (e < E_GCN) atomicAdd(&deg[dst[e]], 1.0f);
}

__global__ void dinv_kernel(const float* __restrict__ deg, float* __restrict__ dinv)
{
    int i = blockIdx.x * blockDim.x + threadIdx.x;
    if (i < N_TOT) dinv[i] = rsqrtf(deg[i]);
}

__global__ void gcn_init_kernel(const float* __restrict__ xw, const float* __restrict__ dinv,
                                const float* __restrict__ bias, float* __restrict__ h)
{
    int t = blockIdx.x * blockDim.x + threadIdx.x;
    if (t >= N_TOT * 64) return;
    int i = t >> 6, f = t & 63;
    float dv = dinv[i];
    h[t] = bias[f] + xw[t] * dv * dv;
}

__global__ void gcn_scatter_kernel(const int* __restrict__ src, const int* __restrict__ dst,
                                   const float* __restrict__ xw, const float* __restrict__ dinv,
                                   float* __restrict__ h)
{
    int t = blockIdx.x * blockDim.x + threadIdx.x;
    if (t >= E_GCN * 64) return;
    int e = t >> 6, f = t & 63;
    int s = src[e], d = dst[e];
    float norm = dinv[s] * dinv[d];
    atomicAdd(&h[(size_t)d * 64 + f], xw[(size_t)s * 64 + f] * norm);
}

__global__ void relu_kernel(float* __restrict__ h, int n)
{
    int t = blockIdx.x * blockDim.x + threadIdx.x;
    if (t < n) h[t] = fmaxf(h[t], 0.0f);
}

// ---------------------------------------------------------------------------
// 600k edge-pair scores: gather over the L2-resident 804x64 table.
// global_prefetch pulls both rows toward the WGP before the dependent loads.
// ---------------------------------------------------------------------------
__global__ void score_kernel(const float* __restrict__ xf, const int* __restrict__ eli,
                             float* __restrict__ out)
{
    int t = blockIdx.x * blockDim.x + threadIdx.x;
    if (t >= E_LBL) return;
    int u = eli[t], v = eli[E_LBL + t];
    const float4* pu = reinterpret_cast<const float4*>(xf + (size_t)u * 64);
    const float4* pv = reinterpret_cast<const float4*>(xf + (size_t)v * 64);
    __builtin_prefetch(pu, 0, 3);
    __builtin_prefetch(pv, 0, 3);
    float acc = 0.0f;
    #pragma unroll
    for (int i = 0; i < 16; ++i) {
        float4 a = pu[i], b = pv[i];
        acc += a.x * b.x + a.y * b.y + a.z * b.z + a.w * b.w;
    }
    out[t] = acc;
}

// ---------------------------------------------------------------------------
// MLP read-out heads: sigmoid(relu(x[idx] @ W1 + b1) @ W2 + b2)
// ---------------------------------------------------------------------------
__global__ void mlp_head_kernel(const float* __restrict__ xf, const int* __restrict__ idx,
                                const float* __restrict__ w1, const float* __restrict__ b1,
                                const float* __restrict__ w2, const float* __restrict__ b2,
                                float* __restrict__ out, int Dout)
{
    __shared__ float s_x[64];
    __shared__ float s_h[64];
    int t = threadIdx.x;
    int r = blockIdx.x;
    s_x[t] = xf[(size_t)idx[r] * 64 + t];
    __syncthreads();
    float acc = b1[t];
    #pragma unroll 8
    for (int i = 0; i < 64; ++i) acc += s_x[i] * w1[i * 64 + t];
    s_h[t] = fmaxf(acc, 0.0f);
    __syncthreads();
    if (t < Dout) {
        float o = b2[t];
        #pragma unroll 8
        for (int i = 0; i < 64; ++i) o += s_h[i] * w2[i * Dout + t];
        out[(size_t)r * Dout + t] = 1.0f / (1.0f + __expf(-o));
    }
}

// ---------------------------------------------------------------------------
extern "C" void kernel_launch(void* const* d_in, const int* in_sizes, int n_in,
                              void* d_out, int out_size, void* d_ws, size_t ws_size,
                              hipStream_t stream)
{
    (void)in_sizes; (void)n_in; (void)out_size; (void)ws_size;

    const float* x        = (const float*)d_in[0];
    const float* x_lnc    = (const float*)d_in[1];
    const float* x_mi     = (const float*)d_in[2];
    const int*   edge     = (const int*)  d_in[3];   // [2,20000]
    const int*   eli      = (const int*)  d_in[4];   // [2,600000]
    const int*   di_idx   = (const int*)  d_in[5];
    const int*   sub_idx  = (const int*)  d_in[6];
    const float* node_lnc = (const float*)d_in[7];
    const float* node_mi  = (const float*)d_in[8];
    const float* WA       = (const float*)d_in[9];
    const float* aA       = (const float*)d_in[10];
    const float* WA_out   = (const float*)d_in[11];
    const float* aA_out   = (const float*)d_in[12];
    const float* WB       = (const float*)d_in[13];
    const float* aB       = (const float*)d_in[14];
    const float* WB_out   = (const float*)d_in[15];
    const float* aB_out   = (const float*)d_in[16];
    const float* bnA_g    = (const float*)d_in[17];
    const float* bnA_b    = (const float*)d_in[18];
    const float* bnB_g    = (const float*)d_in[19];
    const float* bnB_b    = (const float*)d_in[20];
    const float* lnc1_w   = (const float*)d_in[21];
    const float* lnc1_b   = (const float*)d_in[22];
    const float* lnc2_w   = (const float*)d_in[23];
    const float* lnc2_b   = (const float*)d_in[24];
    const float* mi1_w    = (const float*)d_in[25];
    const float* mi1_b    = (const float*)d_in[26];
    const float* mi2_w    = (const float*)d_in[27];
    const float* mi2_b    = (const float*)d_in[28];
    const float* conv1_w  = (const float*)d_in[29];
    const float* conv1_b  = (const float*)d_in[30];
    const float* conv2_w  = (const float*)d_in[31];
    const float* conv2_b  = (const float*)d_in[32];
    const float* dilin1_w = (const float*)d_in[33];
    const float* dilin1_b = (const float*)d_in[34];
    const float* dilin2_w = (const float*)d_in[35];
    const float* dilin2_b = (const float*)d_in[36];
    const float* sublin1_w= (const float*)d_in[37];
    const float* sublin1_b= (const float*)d_in[38];
    const float* sublin2_w= (const float*)d_in[39];
    const float* sublin2_b= (const float*)d_in[40];

    const int* src = edge;             // edge_index[0]
    const int* dst = edge + E_GCN;     // edge_index[1]

    // ------- workspace carve-up (256B aligned) -------
    char* ws = (char*)d_ws;
    size_t off = 0;
    auto take = [&](size_t b) { size_t c = off; off += (b + 255) & ~(size_t)255; return c; };
    float*     x_new = (float*)    (ws + take((size_t)N_TOT * 100 * sizeof(float)));
    _Float16*  A16   = (_Float16*) (ws + take((size_t)816 * 128 * sizeof(_Float16)));
    _Float16*  B16T  = (_Float16*) (ws + take((size_t)64 * 128 * sizeof(_Float16)));
    float*     xw    = (float*)    (ws + take((size_t)N_TOT * 64 * sizeof(float)));
    float*     deg   = (float*)    (ws + take((size_t)N_TOT * sizeof(float)));
    float*     dinv  = (float*)    (ws + take((size_t)N_TOT * sizeof(float)));
    float*     h1    = (float*)    (ws + take((size_t)N_TOT * 64 * sizeof(float)));
    float*     xfin  = (float*)    (ws + take((size_t)N_TOT * 64 * sizeof(float)));

    float* out_scores = (float*)d_out;                 // [600000]
    float* out_di     = out_scores + E_LBL;            // [400,9]
    float* out_sub    = out_di + 400 * 9;              // [400,6]

    // 1) fused per-sample GATs -> x_new
    gat_lnc_kernel<<<N_LNC, 64, 0, stream>>>(x, x_lnc, node_lnc, WA, aA, WA_out, aA_out,
                                             bnA_g, bnA_b, lnc1_w, lnc1_b, lnc2_w, lnc2_b,
                                             x_new);
    gat_mi_kernel<<<N_MI, 32, 0, stream>>>(x, x_mi, node_mi, WB, aB, WB_out, aB_out,
                                           bnB_g, bnB_b, mi1_w, mi1_b, mi2_w, mi2_b,
                                           x_new);

    // 2) GCN degree normalization
    deg_init_kernel<<<(N_TOT + 255) / 256, 256, 0, stream>>>(deg);
    deg_acc_kernel<<<(E_GCN + 255) / 256, 256, 0, stream>>>(dst, deg);
    dinv_kernel<<<(N_TOT + 255) / 256, 256, 0, stream>>>(deg, dinv);

    // 3) conv1: xw = x_new @ conv1_w  (WMMA f16, K=100 padded to 128)
    {
        const int M = N_TOT, K = 100, Kpad = 128, Mpad = 816, N = 64;
        const int Mtiles = 51, Ntiles = 4;
        stage_a_f16<<<(Mpad * Kpad + 255) / 256, 256, 0, stream>>>(x_new, A16, M, K, Kpad, Mpad);
        stage_bT_f16<<<(N * Kpad + 255) / 256, 256, 0, stream>>>(conv1_w, B16T, K, N, Kpad);
        int tiles = Mtiles * Ntiles;
        wmma_gemm_f16<<<(tiles + 7) / 8, 256, 0, stream>>>(A16, B16T, xw,
                                                           M, N, Kpad, Mtiles, Ntiles);
    }
    gcn_init_kernel<<<(N_TOT * 64 + 255) / 256, 256, 0, stream>>>(xw, dinv, conv1_b, h1);
    gcn_scatter_kernel<<<(E_GCN * 64 + 255) / 256, 256, 0, stream>>>(src, dst, xw, dinv, h1);
    relu_kernel<<<(N_TOT * 64 + 255) / 256, 256, 0, stream>>>(h1, N_TOT * 64);

    // 4) conv2: xw = h1 @ conv2_w  (WMMA f16, K=64)
    {
        const int M = N_TOT, K = 64, Kpad = 64, Mpad = 816, N = 64;
        const int Mtiles = 51, Ntiles = 4;
        stage_a_f16<<<(Mpad * Kpad + 255) / 256, 256, 0, stream>>>(h1, A16, M, K, Kpad, Mpad);
        stage_bT_f16<<<(N * Kpad + 255) / 256, 256, 0, stream>>>(conv2_w, B16T, K, N, Kpad);
        int tiles = Mtiles * Ntiles;
        wmma_gemm_f16<<<(tiles + 7) / 8, 256, 0, stream>>>(A16, B16T, xw,
                                                           M, N, Kpad, Mtiles, Ntiles);
    }
    gcn_init_kernel<<<(N_TOT * 64 + 255) / 256, 256, 0, stream>>>(xw, dinv, conv2_b, xfin);
    gcn_scatter_kernel<<<(E_GCN * 64 + 255) / 256, 256, 0, stream>>>(src, dst, xw, dinv, xfin);

    // 5) read-outs
    score_kernel<<<(E_LBL + 255) / 256, 256, 0, stream>>>(xfin, eli, out_scores);
    mlp_head_kernel<<<400, 64, 0, stream>>>(xfin, di_idx, dilin1_w, dilin1_b,
                                            dilin2_w, dilin2_b, out_di, 9);
    mlp_head_kernel<<<400, 64, 0, stream>>>(xfin, sub_idx, sublin1_w, sublin1_b,
                                            sublin2_w, sublin2_b, out_sub, 6);
}
